// PPGNN_498216206705
// MI455X (gfx1250) — matrix operand order repaired and checked by reading, not compiled, not run on hardware
//
#include <hip/hip_runtime.h>
#include <math.h>

#define NN   50000
#define EE   800000
#define CIN  128
#define HH   96
#define COUT 40
#define NL   5
#define DTC  0.05f

typedef __attribute__((ext_vector_type(16))) _Float16 v16h;
typedef __attribute__((ext_vector_type(8)))  _Float16 v8h;
typedef __attribute__((ext_vector_type(8)))  float    v8f;

// branch-free tanh: tanh(z) = 1 - 2/(exp(2z)+1)  (inputs are O(1) here)
__device__ __forceinline__ float fast_tanh(float z) {
    return 1.0f - 2.0f / (__expf(2.0f * z) + 1.0f);
}

// ---------------- CSR build ----------------

__global__ void k_zero_i32(int* p, int n) {
    int i = blockIdx.x * blockDim.x + threadIdx.x;
    if (i < n) p[i] = 0;
}

__global__ void k_deg(const int* __restrict__ dst, int* __restrict__ cnt) {
    int e = blockIdx.x * blockDim.x + threadIdx.x;
    if (e < EE) atomicAdd(&cnt[dst[e]], 1);
}

// single-block exclusive scan over degrees -> row ptr + deg_inv
__global__ void k_scan(const int* __restrict__ deg, int* __restrict__ ptr,
                       float* __restrict__ dinv) {
    __shared__ int ssum[1024];
    int tid = threadIdx.x;
    const int CH = (NN + 1023) / 1024;   // 49
    int b0 = tid * CH, b1 = b0 + CH;
    if (b0 > NN) b0 = NN;
    if (b1 > NN) b1 = NN;
    int s = 0;
    for (int i = b0; i < b1; ++i) s += deg[i];
    ssum[tid] = s;
    __syncthreads();
    for (int off = 1; off < 1024; off <<= 1) {
        int v = (tid >= off) ? ssum[tid - off] : 0;
        __syncthreads();
        ssum[tid] += v;
        __syncthreads();
    }
    int run = (tid == 0) ? 0 : ssum[tid - 1];
    for (int i = b0; i < b1; ++i) {
        ptr[i] = run;
        int d = deg[i];
        dinv[i] = 1.0f / fmaxf((float)d, 1.0f);
        run += d;
    }
    if (tid == 1023) ptr[NN] = run;
}

__global__ void k_fill(const int* __restrict__ src, const int* __restrict__ dst,
                       const int* __restrict__ ptr, int* __restrict__ cnt,
                       int* __restrict__ csr) {
    int e = blockIdx.x * blockDim.x + threadIdx.x;
    if (e < EE) {
        int d = dst[e];
        int pos = ptr[d] + atomicAdd(&cnt[d], 1);
        csr[pos] = src[e];
    }
}

// ---------------- one-time conversions / repacks ----------------

// f32 -> f16 elementwise (pairs, so stores merge to b32)
__global__ void k_f32_to_f16(const float* __restrict__ in, _Float16* __restrict__ o,
                             int n2) {
    int i = blockIdx.x * blockDim.x + threadIdx.x;
    if (i < n2) {
        float2 v = ((const float2*)in)[i];
        o[2 * i]     = (_Float16)v.x;
        o[2 * i + 1] = (_Float16)v.y;
    }
}

// repack W [Ktot x ncols] (row-major f32) into f16 B-fragments:
// out[((kbi*colTiles + ct)*32 + lane) * 16 + 2v+j] = W[(kb+16h+2v+j)*ncols + ct*16+m]
// zero-padded when ncol >= ncols (readout's 40 -> 48 padding)
__global__ void k_packB(const float* __restrict__ W, _Float16* __restrict__ out,
                        int Ktot, int ncols, int colTiles) {
    int t = blockIdx.x * blockDim.x + threadIdx.x;
    int total = (Ktot / 32) * colTiles * 32;
    if (t >= total) return;
    int lane = t & 31;
    int ct   = (t >> 5) % colTiles;
    int kbi  = t / (32 * colTiles);
    int kb = kbi * 32;
    int m = lane & 15, h = lane >> 4;
    int ncol = ct * 16 + m;
    bool cok = ncol < ncols;
    _Float16* o = out + (size_t)t * 16;
#pragma unroll
    for (int v = 0; v < 8; ++v) {
        int k = kb + 16 * h + 2 * v;
        float a0 = cok ? W[(size_t)k * ncols + ncol]       : 0.0f;
        float a1 = cok ? W[(size_t)(k + 1) * ncols + ncol] : 0.0f;
        o[2 * v]     = (_Float16)a0;
        o[2 * v + 1] = (_Float16)a1;
    }
}

// concat [X|Y] (f32, N x 96 each) -> f16 [N x 192]
__global__ void k_concat16(const float* __restrict__ X, const float* __restrict__ Y,
                           _Float16* __restrict__ xy) {
    int i = blockIdx.x * blockDim.x + threadIdx.x;
    if (i < NN * 2 * HH) {
        int node = i / (2 * HH);
        int c    = i % (2 * HH);
        float v = (c < HH) ? X[(size_t)node * HH + c] : Y[(size_t)node * HH + (c - HH)];
        xy[i] = (_Float16)v;
    }
}

// ---------------- Lift: X = tanh(x@Wx+bx), Y = tanh(x@Wy+by) via WMMA ----------------
// one wave per (16-row tile, species); A frag = 2 contiguous 16B f16 loads,
// B frag = 1 contiguous 32B pre-swizzled f16 load; A reused across 6 column tiles.

__global__ void k_lift(const _Float16* __restrict__ x16,
                       const _Float16* __restrict__ Wxs, const float* __restrict__ bx,
                       const _Float16* __restrict__ Wys, const float* __restrict__ by,
                       float* __restrict__ X, float* __restrict__ Y) {
    const int ROWT = NN / 16;   // 3125
    const int COLT = HH / 16;   // 6
    int wid = blockIdx.x * (blockDim.x >> 5) + (threadIdx.x >> 5);
    if (wid >= ROWT * 2) return;            // wave-uniform: EXEC stays all-ones
    int sp = wid & 1;
    int rt = wid >> 1;
    const _Float16* Ws   = sp ? Wys : Wxs;
    const float*    bias = sp ? by  : bx;
    float*          Out  = sp ? Y   : X;

    int lane = threadIdx.x & 31;
    int m = lane & 15, h = lane >> 4;
    int row = rt * 16 + m;                  // A row (node)
    const _Float16* xr = x16 + (size_t)row * CIN;

    v8f acc[COLT] = {};
#pragma unroll
    for (int kbi = 0; kbi < CIN / 32; ++kbi) {
        int kb = kbi * 32;
        // A frag: elements 0..15 -> K = kb+16h+{0..15}
        v8h alo = *(const v8h*)(xr + kb + 16 * h);
        v8h ahi = *(const v8h*)(xr + kb + 16 * h + 8);
        v16h a = __builtin_shufflevector(alo, ahi, 0, 1, 2, 3, 4, 5, 6, 7,
                                         8, 9, 10, 11, 12, 13, 14, 15);
#pragma unroll
        for (int ct = 0; ct < COLT; ++ct) {
            v16h bf = *(const v16h*)(Ws + ((size_t)(kbi * COLT + ct) * 32 + lane) * 16);
            acc[ct] = __builtin_amdgcn_wmma_f32_16x16x32_f16(false, a, false, bf,
                                                             (short)0, acc[ct],
                                                             false, false);
        }
    }
#pragma unroll
    for (int ct = 0; ct < COLT; ++ct) {
        int ncol = ct * 16 + m;
        float bv = bias[ncol];
#pragma unroll
        for (int r = 0; r < 8; ++r) {       // D VGPR r -> M = r + 8h
            int mm = rt * 16 + r + 8 * h;
            Out[(size_t)mm * HH + ncol] = fast_tanh(acc[ct][r] + bv);
        }
    }
}

// ---------------- Fused mean-aggregation + LV Euler step ----------------
// one wave per node; lane owns channels 3*lane .. 3*lane+2 (contiguous -> b96 loads)

__global__ void k_layer(const float* __restrict__ X, const float* __restrict__ Y,
                        float* __restrict__ Xn, float* __restrict__ Yn,
                        const int* __restrict__ ptr, const int* __restrict__ csr,
                        const float* __restrict__ dinv,
                        const float* __restrict__ al, const float* __restrict__ be,
                        const float* __restrict__ ga, const float* __restrict__ de) {
    int node = blockIdx.x * (blockDim.x >> 5) + (threadIdx.x >> 5);
    if (node >= NN) return;
    int lane = threadIdx.x & 31;
    int c0 = 3 * lane;
    float sx0 = 0.f, sx1 = 0.f, sx2 = 0.f, sy0 = 0.f, sy1 = 0.f, sy2 = 0.f;
    int b = ptr[node], e = ptr[node + 1];
    for (int i = b; i < e; ++i) {
        int s = csr[i];
        if (i + 1 < e) {                     // hide L2 gather latency
            int sn = csr[i + 1];
            __builtin_prefetch(X + (size_t)sn * HH + c0, 0, 0);
            __builtin_prefetch(Y + (size_t)sn * HH + c0, 0, 0);
        }
        const float* xr = X + (size_t)s * HH + c0;
        const float* yr = Y + (size_t)s * HH + c0;
        sx0 += xr[0]; sx1 += xr[1]; sx2 += xr[2];
        sy0 += yr[0]; sy1 += yr[1]; sy2 += yr[2];
    }
    float di = dinv[node];
    const float* xr = X + (size_t)node * HH + c0;
    const float* yr = Y + (size_t)node * HH + c0;
    float* xo = Xn + (size_t)node * HH + c0;
    float* yo = Yn + (size_t)node * HH + c0;
#pragma unroll
    for (int j = 0; j < 3; ++j) {
        int c = c0 + j;
        float ax = (j == 0 ? sx0 : j == 1 ? sx1 : sx2) * di;
        float ay = (j == 0 ? sy0 : j == 1 ? sy1 : sy2) * di;
        float xv = xr[j], yv = yr[j];
        xo[j] = xv + DTC * xv * (al[c] - be[c] * ay);
        yo[j] = yv + DTC * yv * (-ga[c] + de[c] * ax);
    }
}

// ---------------- Readout: out = [X,Y] @ Wr + br via WMMA ----------------
// one wave per 16-row tile; A from f16 concat buffer, B pre-swizzled+padded.

__global__ void k_readout(const _Float16* __restrict__ xy16,
                          const _Float16* __restrict__ Wrs,
                          const float* __restrict__ br,
                          float* __restrict__ out) {
    const int ROWT = NN / 16;   // 3125
    const int COLT = 3;         // 48 cols, last 8 zero-padded (COUT=40)
    const int KT   = 2 * HH;    // 192
    int wid = blockIdx.x * (blockDim.x >> 5) + (threadIdx.x >> 5);
    if (wid >= ROWT) return;
    int rt = wid;
    int lane = threadIdx.x & 31;
    int m = lane & 15, h = lane >> 4;
    int row = rt * 16 + m;
    const _Float16* xr = xy16 + (size_t)row * KT;

    v8f acc[COLT] = {};
#pragma unroll
    for (int kbi = 0; kbi < KT / 32; ++kbi) {
        int kb = kbi * 32;
        v8h alo = *(const v8h*)(xr + kb + 16 * h);
        v8h ahi = *(const v8h*)(xr + kb + 16 * h + 8);
        v16h a = __builtin_shufflevector(alo, ahi, 0, 1, 2, 3, 4, 5, 6, 7,
                                         8, 9, 10, 11, 12, 13, 14, 15);
#pragma unroll
        for (int ct = 0; ct < COLT; ++ct) {
            v16h bf = *(const v16h*)(Wrs + ((size_t)(kbi * COLT + ct) * 32 + lane) * 16);
            acc[ct] = __builtin_amdgcn_wmma_f32_16x16x32_f16(false, a, false, bf,
                                                             (short)0, acc[ct],
                                                             false, false);
        }
    }
#pragma unroll
    for (int ct = 0; ct < COLT; ++ct) {
        int ncol = ct * 16 + m;
        if (ncol < COUT) {
            float bv = br[ncol];
#pragma unroll
            for (int r = 0; r < 8; ++r) {
                int mm = rt * 16 + r + 8 * h;
                out[(size_t)mm * COUT + ncol] = acc[ct][r] + bv;
            }
        }
    }
}

// ---------------- host launcher ----------------

extern "C" void kernel_launch(void* const* d_in, const int* in_sizes, int n_in,
                              void* d_out, int out_size, void* d_ws, size_t ws_size,
                              hipStream_t stream) {
    (void)in_sizes; (void)n_in; (void)out_size; (void)ws_size;
    const float* x     = (const float*)d_in[0];
    const int*   eidx  = (const int*)d_in[1];
    const float* Wx    = (const float*)d_in[2];
    const float* bx    = (const float*)d_in[3];
    const float* Wy    = (const float*)d_in[4];
    const float* by    = (const float*)d_in[5];
    const float* alpha = (const float*)d_in[6];
    const float* beta  = (const float*)d_in[7];
    const float* gamma = (const float*)d_in[8];
    const float* delta = (const float*)d_in[9];
    const float* Wr    = (const float*)d_in[10];
    const float* br    = (const float*)d_in[11];
    float* out = (float*)d_out;
    const int* src = eidx;        // edge_index[0]
    const int* dst = eidx + EE;   // edge_index[1]

    char* p = (char*)d_ws;
    auto wsalloc = [&](size_t bytes) -> char* {
        char* r = p;
        p += (bytes + 255) & ~(size_t)255;
        return r;
    };
    float*     dinv = (float*)wsalloc((size_t)NN * 4);
    int*       ptr  = (int*)  wsalloc((size_t)(NN + 1) * 4);
    int*       cnt  = (int*)  wsalloc((size_t)NN * 4);
    int*       csr  = (int*)  wsalloc((size_t)EE * 4);
    float*     Xa   = (float*)wsalloc((size_t)NN * HH * 4);
    float*     Ya   = (float*)wsalloc((size_t)NN * HH * 4);
    float*     Xb   = (float*)wsalloc((size_t)NN * HH * 4);
    float*     Yb   = (float*)wsalloc((size_t)NN * HH * 4);
    _Float16*  x16  = (_Float16*)wsalloc((size_t)NN * CIN * 2);
    _Float16*  xy16 = (_Float16*)wsalloc((size_t)NN * 2 * HH * 2);
    _Float16*  Wxs  = (_Float16*)wsalloc((size_t)(CIN / 32) * (HH / 16) * 32 * 16 * 2);
    _Float16*  Wys  = (_Float16*)wsalloc((size_t)(CIN / 32) * (HH / 16) * 32 * 16 * 2);
    _Float16*  Wrs  = (_Float16*)wsalloc((size_t)(2 * HH / 32) * 3 * 32 * 16 * 2);

    // one-time conversions / repacks
    int xn2 = NN * CIN / 2;
    k_f32_to_f16<<<(xn2 + 255) / 256, 256, 0, stream>>>(x, x16, xn2);
    int packLift = (CIN / 32) * (HH / 16) * 32;      // 768
    k_packB<<<(packLift + 255) / 256, 256, 0, stream>>>(Wx, Wxs, CIN, HH, HH / 16);
    k_packB<<<(packLift + 255) / 256, 256, 0, stream>>>(Wy, Wys, CIN, HH, HH / 16);
    int packRo = (2 * HH / 32) * 3 * 32;             // 576
    k_packB<<<(packRo + 255) / 256, 256, 0, stream>>>(Wr, Wrs, 2 * HH, COUT, 3);

    // CSR build (degree -> scan -> bucket fill)
    k_zero_i32<<<(NN + 255) / 256, 256, 0, stream>>>(cnt, NN);
    k_deg<<<(EE + 255) / 256, 256, 0, stream>>>(dst, cnt);
    k_scan<<<1, 1024, 0, stream>>>(cnt, ptr, dinv);
    k_zero_i32<<<(NN + 255) / 256, 256, 0, stream>>>(cnt, NN);
    k_fill<<<(EE + 255) / 256, 256, 0, stream>>>(src, dst, ptr, cnt, csr);

    // Lift (WMMA): 3125 row-tiles x 2 species, 8 waves / block
    int liftWaves = (NN / 16) * 2;
    k_lift<<<(liftWaves + 7) / 8, 256, 0, stream>>>(x16, Wxs, bx, Wys, by, Xa, Ya);

    // 5 LV layers, ping-pong buffers
    float* Xc = Xa; float* Yc = Ya; float* Xo = Xb; float* Yo = Yb;
    for (int l = 0; l < NL; ++l) {
        k_layer<<<(NN + 7) / 8, 256, 0, stream>>>(Xc, Yc, Xo, Yo, ptr, csr, dinv,
                                                  alpha + l * HH, beta + l * HH,
                                                  gamma + l * HH, delta + l * HH);
        float* t;
        t = Xc; Xc = Xo; Xo = t;
        t = Yc; Yc = Yo; Yo = t;
    }

    // final features -> f16 concat for readout A path
    int cn = NN * 2 * HH;
    k_concat16<<<(cn + 255) / 256, 256, 0, stream>>>(Xc, Yc, xy16);

    // Readout (WMMA): 3125 row-tiles (40 cols padded to 48 in packed B)
    int roWaves = NN / 16;
    k_readout<<<(roWaves + 7) / 8, 256, 0, stream>>>(xy16, Wrs, br, out);
}
